// Volcano_31920196944557
// MI455X (gfx1250) — matrix operand order repaired and checked by reading, not compiled, not run on hardware
//
#include <hip/hip_runtime.h>
#include <stdint.h>
#include <stddef.h>

// ---------------- problem constants ----------------
constexpr int Nb = 64, Sx = 256, Lx = 1024;
constexpr int Hx = 512, Ax = 512, Bx = 512, Ex = 1024, Vx = 34;
constexpr int EB = Ex + Bx;   // 1536
constexpr int G4 = 4 * Hx;    // 2048

// ---------------- launch geometry ------------------
#define NWG   64
#define NTHR  256
#define NT    (NWG * NTHR)     // 16384 threads
#define NWAVES (NT / 32)       // 512 wave32s
static_assert(NWG == Nb, "softmax assigns one workgroup per batch row");

typedef __attribute__((ext_vector_type(16))) __bf16 v16bf;
typedef __attribute__((ext_vector_type(8)))  __bf16 v8bf;
typedef __attribute__((ext_vector_type(8)))  float  v8f;

__device__ __forceinline__ uint16_t f2bf(float f) {
  uint32_t u = __float_as_uint(f);
  uint32_t r = (u >> 16) & 1u;
  return (uint16_t)((u + 0x7fffu + r) >> 16);
}
__device__ __forceinline__ float bf2f(uint16_t h) {
  return __uint_as_float(((uint32_t)h) << 16);
}
__device__ __forceinline__ float sigm(float x) { return 1.0f / (1.0f + __expf(-x)); }

// ---------------- WMMA fragment loaders ----------------
// A (16x32 bf16, row-major src, ld = K): lane<16 -> row=lane, K=[k..k+7],[k+16..k+23]
//                                        lane>=16 -> row=lane-16, K=[k+8..k+15],[k+24..k+31]
__device__ __forceinline__ v16bf load_a_frag(const uint16_t* rowbase, int ld, int kk, int lane) {
  const __bf16* p = reinterpret_cast<const __bf16*>(rowbase)
                  + (size_t)(lane & 15) * ld + kk + ((lane & 16) ? 8 : 0);
  v8bf lo = *reinterpret_cast<const v8bf*>(p);
  v8bf hi = *reinterpret_cast<const v8bf*>(p + 16);
  return __builtin_shufflevector(lo, hi, 0,1,2,3,4,5,6,7,8,9,10,11,12,13,14,15);
}
// B (32x16 bf16) from weight W stored [Nout][K] row-major (i.e. already B^T source):
// lane<16 -> col=n0+lane, K=[k..k+7],[k+16..k+23]; lane>=16 -> col=n0+lane-16, K=[k+8..k+15],[k+24..k+31]
__device__ __forceinline__ v16bf load_b_frag(const uint16_t* w, int ld, int n0, int kk, int lane) {
  const __bf16* p = reinterpret_cast<const __bf16*>(w)
                  + (size_t)(n0 + (lane & 15)) * ld + kk + ((lane & 16) ? 8 : 0);
  v8bf lo = *reinterpret_cast<const v8bf*>(p);
  v8bf hi = *reinterpret_cast<const v8bf*>(p + 16);
  return __builtin_shufflevector(lo, hi, 0,1,2,3,4,5,6,7,8,9,10,11,12,13,14,15);
}

// out[M][Nout] = A1[M][K1] * W1[Nout][K1]^T (+ A2[M][K2] * W2[Nout][K2]^T) + b1 (+ b2)
__device__ void wmma_gemm(const uint16_t* A1, int K1, const uint16_t* W1,
                          const uint16_t* A2, int K2, const uint16_t* W2,
                          const float* b1, const float* b2,
                          float* out, int M, int Nout, int waveId, int lane)
{
  const int ntc = Nout >> 4;
  const int ntiles = (M >> 4) * ntc;
  for (int t = waveId; t < ntiles; t += NWAVES) {
    const int m0 = (t / ntc) << 4;
    const int n0 = (t % ntc) << 4;
    v8f acc = {};
    const uint16_t* arow = A1 + (size_t)m0 * K1;
    for (int kk = 0; kk < K1; kk += 32) {
      v16bf a = load_a_frag(arow, K1, kk, lane);
      v16bf b = load_b_frag(W1, K1, n0, kk, lane);
      acc = __builtin_amdgcn_wmma_f32_16x16x32_bf16(false, a, false, b, (short)0, acc,
                                                    false, false);
    }
    if (A2) {
      const uint16_t* arow2 = A2 + (size_t)m0 * K2;
      for (int kk = 0; kk < K2; kk += 32) {
        v16bf a = load_a_frag(arow2, K2, kk, lane);
        v16bf b = load_b_frag(W2, K2, n0, kk, lane);
        acc = __builtin_amdgcn_wmma_f32_16x16x32_bf16(false, a, false, b, (short)0, acc,
                                                      false, false);
      }
    }
    const int col = n0 + (lane & 15);
    float bias = (b1 ? b1[col] : 0.0f) + (b2 ? b2[col] : 0.0f);
    const int r0 = m0 + ((lane & 16) ? 8 : 0);
    float* orow = out + (size_t)r0 * Nout + col;
#pragma unroll
    for (int vv = 0; vv < 8; ++vv) orow[(size_t)vv * Nout] = acc[vv] + bias;
  }
}

// ---------------- grid-wide barrier ----------------
__device__ __forceinline__ void gbar(unsigned* cnt, unsigned* gen, unsigned& lg) {
  __syncthreads();
  if (threadIdx.x == 0) {
    __threadfence();
    const unsigned target = lg + 1u;
    if (atomicAdd(cnt, 1u) == (unsigned)(NWG - 1)) {
      atomicExch(cnt, 0u);
      __threadfence();
      __hip_atomic_store(gen, target, __ATOMIC_RELEASE, __HIP_MEMORY_SCOPE_AGENT);
    } else {
      while (__hip_atomic_load(gen, __ATOMIC_ACQUIRE, __HIP_MEMORY_SCOPE_AGENT) < target)
        __builtin_amdgcn_s_sleep(1);
    }
  }
  __syncthreads();
  lg++;
}

// ---------------- parameter block ----------------
struct Params {
  const int*   transcripts;            // [N,S]
  const int*   alen;                   // [N]
  const float* keys;                   // [N,L,A] f32
  const float* values;                 // [N,L,B] f32
  const float* W_out;                  // [V,E]
  const float* b_out;                  // [V]
  const float* init_ctx;               // [B]
  const float* h_init[3];
  const float* c_init[3];
  const float* bih[3];
  const float* bhh[3];
  const float* bq;
  float*       out;                    // [N,S,V]
  unsigned*    cnt;
  unsigned*    gen;
  const uint16_t* WihB[3];             // bf16 weights
  const uint16_t* WhhB[3];
  const uint16_t* WqB;
  uint16_t* xbuf;                      // [N,EB] bf16
  float*    gates;                     // [N,4H]
  uint16_t* hB[3];                     // [N,H] bf16
  float*    cst[3];                    // [N,H]
  float*    qf;                        // [N,A]
  float*    energy;                    // [N,L]
  float*    att;                       // [N,L]
  float*    ctxv;                      // [N,B]
  const uint16_t* keysB;               // optional bf16 KV (L2-resident), may be null
  const uint16_t* valsB;
};

// ---------------- f32 -> bf16 converter ----------------
__global__ void cvt_bf16_kernel(const float* __restrict__ in, uint16_t* __restrict__ out,
                                size_t n) {
  size_t i = (size_t)blockIdx.x * blockDim.x + threadIdx.x;
  const size_t stride = (size_t)gridDim.x * blockDim.x;
  for (; i < n; i += stride) out[i] = f2bf(in[i]);
}

// ---------------- persistent decoder ----------------
__global__ __launch_bounds__(NTHR) void las_persistent(Params p) {
  const int tid    = threadIdx.x;
  const int gtid   = blockIdx.x * NTHR + tid;
  const int lane   = tid & 31;
  const int waveId = gtid >> 5;
  unsigned lg = 0;
  __shared__ float red[NTHR];

  // ---- init recurrent state ----
  for (int i = gtid; i < Nb * Hx; i += NT) {
    const int j = i % Hx;
#pragma unroll
    for (int l = 0; l < 3; ++l) {
      p.cst[l][i] = p.c_init[l][j];
      p.hB[l][i]  = f2bf(p.h_init[l][j]);
    }
  }
  for (int i = gtid; i < Nb * Bx; i += NT) p.ctxv[i] = p.init_ctx[i % Bx];
  gbar(p.cnt, p.gen, lg);

  for (int t = 0; t < Sx; ++t) {
    // ---- phase 1: x = [embed(tok) | ctx] in bf16 ----
    for (int i = gtid; i < Nb * EB; i += NT) {
      const int n = i / EB, e = i % EB;
      float v;
      if (e < Ex) {
        const int tok = p.transcripts[n * Sx + t];
        v = p.W_out[(size_t)tok * Ex + e];
      } else {
        v = p.ctxv[n * Bx + (e - Ex)];
      }
      p.xbuf[i] = f2bf(v);
    }
    gbar(p.cnt, p.gen, lg);

    // ---- three LSTM layers: WMMA gates + VALU cell update ----
#pragma unroll 1
    for (int l = 0; l < 3; ++l) {
      if (l == 0)
        wmma_gemm(p.xbuf, EB, p.WihB[0], p.hB[0], Hx, p.WhhB[0],
                  p.bih[0], p.bhh[0], p.gates, Nb, G4, waveId, lane);
      else
        wmma_gemm(p.hB[l - 1], Hx, p.WihB[l], p.hB[l], Hx, p.WhhB[l],
                  p.bih[l], p.bhh[l], p.gates, Nb, G4, waveId, lane);
      gbar(p.cnt, p.gen, lg);

      for (int i = gtid; i < Nb * Hx; i += NT) {
        const int n = i / Hx, j = i % Hx;
        const float* g = p.gates + (size_t)n * G4;
        const float cc = sigm(g[Hx + j]) * p.cst[l][i] + sigm(g[j]) * tanhf(g[2 * Hx + j]);
        p.cst[l][i] = cc;
        p.hB[l][i]  = f2bf(sigm(g[3 * Hx + j]) * tanhf(cc));
      }
      gbar(p.cnt, p.gen, lg);
    }

    // ---- query projection (WMMA) ----
    wmma_gemm(p.hB[2], Hx, p.WqB, nullptr, 0, nullptr,
              p.bq, nullptr, p.qf, Nb, Ax, waveId, lane);
    gbar(p.cnt, p.gen, lg);

    // ---- energy: e[n,l] = q[n] . keys[n,l]  (batched GEMV, load-bound) ----
    for (int i = gtid; i < Nb * Lx; i += NT) {
      const int n = i / Lx, l = i % Lx;
      float e = 0.0f;
      const float* qp = p.qf + (size_t)n * Ax;
      if (p.keysB) {
        const uint16_t* kp = p.keysB + ((size_t)n * Lx + l) * Ax;
        for (int a = 0; a < Ax; a += 4)
          e += qp[a] * bf2f(kp[a]) + qp[a + 1] * bf2f(kp[a + 1])
             + qp[a + 2] * bf2f(kp[a + 2]) + qp[a + 3] * bf2f(kp[a + 3]);
        __builtin_prefetch(p.valsB + ((size_t)n * Lx + l) * Bx, 0, 1);
      } else {
        const float4* kp = reinterpret_cast<const float4*>(p.keys + ((size_t)n * Lx + l) * Ax);
        const float4* q4 = reinterpret_cast<const float4*>(qp);
        for (int a = 0; a < Ax / 4; ++a) {
          const float4 k = kp[a], q = q4[a];
          e += k.x * q.x + k.y * q.y + k.z * q.z + k.w * q.w;
        }
        __builtin_prefetch(p.values + ((size_t)n * Lx + l) * Bx, 0, 1);
      }
      p.energy[i] = (l < p.alen[n]) ? e : 0.0f;
    }
    gbar(p.cnt, p.gen, lg);

    // ---- masked softmax, one workgroup per batch row (the 1/s1 factor cancels
    //      under the reference's renormalization, so only max + masked-sum) ----
    {
      const int n  = blockIdx.x;
      const int al = p.alen[n];
      float m = -1e30f;
      for (int l = tid; l < Lx; l += NTHR) m = fmaxf(m, p.energy[n * Lx + l]);
      red[tid] = m; __syncthreads();
      for (int s = NTHR / 2; s > 0; s >>= 1) {
        if (tid < s) red[tid] = fmaxf(red[tid], red[tid + s]);
        __syncthreads();
      }
      const float mx = red[0]; __syncthreads();
      float sum = 0.0f;
      for (int l = tid; l < Lx; l += NTHR) {
        const float pv = (l < al) ? __expf(p.energy[n * Lx + l] - mx) : 0.0f;
        p.att[n * Lx + l] = pv;
        sum += pv;
      }
      red[tid] = sum; __syncthreads();
      for (int s = NTHR / 2; s > 0; s >>= 1) {
        if (tid < s) red[tid] += red[tid + s];
        __syncthreads();
      }
      const float inv = 1.0f / red[0]; __syncthreads();
      for (int l = tid; l < Lx; l += NTHR) p.att[n * Lx + l] *= inv;
    }
    gbar(p.cnt, p.gen, lg);

    // ---- ctx[n,b] = sum_l att[n,l] * values[n,l,b] (coalesced over b) ----
    for (int i = gtid; i < Nb * Bx; i += NT) {
      const int n = i / Bx, b = i % Bx;
      const float* ap = p.att + (size_t)n * Lx;
      float s = 0.0f;
      if (p.valsB) {
        const uint16_t* vp = p.valsB + (size_t)n * Lx * Bx + b;
        for (int l = 0; l < Lx; l += 2)
          s += ap[l] * bf2f(vp[(size_t)l * Bx]) + ap[l + 1] * bf2f(vp[(size_t)(l + 1) * Bx]);
      } else {
        const float* vp = p.values + (size_t)n * Lx * Bx + b;
        for (int l = 0; l < Lx; l += 2)
          s += ap[l] * vp[(size_t)l * Bx] + ap[l + 1] * vp[(size_t)(l + 1) * Bx];
      }
      p.ctxv[i] = s;
    }
    gbar(p.cnt, p.gen, lg);

    // ---- logits[n,v] = [q|ctx] . W_out[v] + b_out[v]  (V=34, tiny) ----
    for (int i = gtid; i < Nb * Vx; i += NT) {
      const int n = i / Vx, vo = i % Vx;
      const float4* w  = reinterpret_cast<const float4*>(p.W_out + (size_t)vo * Ex);
      const float4* q4 = reinterpret_cast<const float4*>(p.qf + (size_t)n * Ax);
      const float4* c4 = reinterpret_cast<const float4*>(p.ctxv + (size_t)n * Bx);
      float s = p.b_out[vo];
      for (int a = 0; a < Ax / 4; ++a) {
        const float4 ww = w[a], qq = q4[a];
        s += ww.x * qq.x + ww.y * qq.y + ww.z * qq.z + ww.w * qq.w;
      }
      for (int b = 0; b < Bx / 4; ++b) {
        const float4 ww = w[Ax / 4 + b], cc = c4[b];
        s += ww.x * cc.x + ww.y * cc.y + ww.z * cc.z + ww.w * cc.w;
      }
      p.out[((size_t)n * Sx + t) * Vx + vo] = s;
    }
    gbar(p.cnt, p.gen, lg);
  }
}

// ---------------- host launch ----------------
extern "C" void kernel_launch(void* const* d_in, const int* in_sizes, int n_in,
                              void* d_out, int out_size, void* d_ws, size_t ws_size,
                              hipStream_t stream) {
  (void)in_sizes; (void)n_in; (void)out_size;

  const int*   transcripts = (const int*)  d_in[0];
  const int*   alen        = (const int*)  d_in[2];
  const float* keys        = (const float*)d_in[3];
  const float* values      = (const float*)d_in[4];
  const float* W_out       = (const float*)d_in[5];
  const float* b_out       = (const float*)d_in[6];
  const float* init_ctx    = (const float*)d_in[7];
  const float* h_init[3]   = {(const float*)d_in[8],  (const float*)d_in[10], (const float*)d_in[12]};
  const float* c_init[3]   = {(const float*)d_in[9],  (const float*)d_in[11], (const float*)d_in[13]};
  const float* Wih[3]      = {(const float*)d_in[14], (const float*)d_in[18], (const float*)d_in[22]};
  const float* Whh[3]      = {(const float*)d_in[15], (const float*)d_in[19], (const float*)d_in[23]};
  const float* bih[3]      = {(const float*)d_in[16], (const float*)d_in[20], (const float*)d_in[24]};
  const float* bhh[3]      = {(const float*)d_in[17], (const float*)d_in[21], (const float*)d_in[25]};
  const float* W_q         = (const float*)d_in[26];
  const float* b_q         = (const float*)d_in[27];

  // ---- carve workspace ----
  uintptr_t base = (uintptr_t)d_ws;
  uintptr_t end  = base + ws_size;
  auto alloc = [&](size_t bytes) -> void* {
    void* r = (void*)base;
    base += (bytes + 255u) & ~(size_t)255u;
    return r;
  };
  unsigned* sync = (unsigned*)alloc(256);                    // [0]=cnt [1]=gen
  uint16_t* WihB[3], *WhhB[3];
  WihB[0] = (uint16_t*)alloc((size_t)G4 * EB * 2);
  WhhB[0] = (uint16_t*)alloc((size_t)G4 * Hx * 2);
  for (int l = 1; l < 3; ++l) {
    WihB[l] = (uint16_t*)alloc((size_t)G4 * Hx * 2);
    WhhB[l] = (uint16_t*)alloc((size_t)G4 * Hx * 2);
  }
  uint16_t* WqB   = (uint16_t*)alloc((size_t)Ax * Hx * 2);
  uint16_t* xbuf  = (uint16_t*)alloc((size_t)Nb * EB * 2);
  float*    gates = (float*)   alloc((size_t)Nb * G4 * 4);
  uint16_t* hB[3]; float* cst[3];
  for (int l = 0; l < 3; ++l) {
    hB[l]  = (uint16_t*)alloc((size_t)Nb * Hx * 2);
    cst[l] = (float*)   alloc((size_t)Nb * Hx * 4);
  }
  float* qf     = (float*)alloc((size_t)Nb * Ax * 4);
  float* energy = (float*)alloc((size_t)Nb * Lx * 4);
  float* att    = (float*)alloc((size_t)Nb * Lx * 4);
  float* ctxv   = (float*)alloc((size_t)Nb * Bx * 4);

  // optional: bf16 keys/values (134 MB total -> fully L2-resident on MI455X)
  const size_t kvElems = (size_t)Nb * Lx * Ax;
  uint16_t* keysB = nullptr; uint16_t* valsB = nullptr;
  if (end - base >= 2u * kvElems * 2u + 512u) {
    keysB = (uint16_t*)alloc(kvElems * 2);
    valsB = (uint16_t*)alloc(kvElems * 2);
  }

  hipMemsetAsync(sync, 0, 8, stream);

  // ---- weight (and KV) conversion to bf16 ----
  cvt_bf16_kernel<<<512, 256, 0, stream>>>(Wih[0], WihB[0], (size_t)G4 * EB);
  cvt_bf16_kernel<<<512, 256, 0, stream>>>(Whh[0], WhhB[0], (size_t)G4 * Hx);
  for (int l = 1; l < 3; ++l) {
    cvt_bf16_kernel<<<512, 256, 0, stream>>>(Wih[l], WihB[l], (size_t)G4 * Hx);
    cvt_bf16_kernel<<<512, 256, 0, stream>>>(Whh[l], WhhB[l], (size_t)G4 * Hx);
  }
  cvt_bf16_kernel<<<512, 256, 0, stream>>>(W_q, WqB, (size_t)Ax * Hx);
  if (keysB) {
    cvt_bf16_kernel<<<2048, 256, 0, stream>>>(keys,   keysB, kvElems);
    cvt_bf16_kernel<<<2048, 256, 0, stream>>>(values, valsB, kvElems);
  }

  // ---- persistent decode ----
  Params p;
  p.transcripts = transcripts; p.alen = alen;
  p.keys = keys; p.values = values;
  p.W_out = W_out; p.b_out = b_out; p.init_ctx = init_ctx;
  for (int l = 0; l < 3; ++l) {
    p.h_init[l] = h_init[l]; p.c_init[l] = c_init[l];
    p.bih[l] = bih[l];       p.bhh[l] = bhh[l];
    p.WihB[l] = WihB[l];     p.WhhB[l] = WhhB[l];
    p.hB[l] = hB[l];         p.cst[l] = cst[l];
  }
  p.bq = b_q; p.out = (float*)d_out;
  p.cnt = sync; p.gen = sync + 1;
  p.WqB = WqB; p.xbuf = xbuf; p.gates = gates;
  p.qf = qf; p.energy = energy; p.att = att; p.ctxv = ctxv;
  p.keysB = keysB; p.valsB = valsB;

  las_persistent<<<NWG, NTHR, 0, stream>>>(p);
}
